// Intra_Attention_withZeromap_19464791785476
// MI455X (gfx1250) — matrix operand order repaired and checked by reading, not compiled, not run on hardware
//
#include <hip/hip_runtime.h>

typedef __attribute__((ext_vector_type(16))) _Float16 v16h;
typedef __attribute__((ext_vector_type(8)))  _Float16 v8h;
typedef __attribute__((ext_vector_type(8)))  float    v8f;

#define HEADS 4
#define WIMG  384
#define HWPIX (384*384)          // 147456
#define BATCH 2
#define PTOT  (BATCH*HWPIX)      // 294912 pixels
#define CCH   64
#define TT    64                 // tokens per 8x8 block group
#define FF    2304               // 48*48 blocks
#define FC    128                // f-chunk staged in LDS
#define NCHUNK (FF / FC)         // 18

__device__ __forceinline__ v16h combine16(v8h lo, v8h hi) {
  return __builtin_shufflevector(lo, hi, 0,1,2,3,4,5,6,7,8,9,10,11,12,13,14,15);
}

__device__ __forceinline__ v8f wmma32(v16h a, v16h b, v8f c) {
  return __builtin_amdgcn_wmma_f32_16x16x32_f16(false, a, false, b, (short)0, c,
                                                false, false);
}

// A fragment (16x32 f16) from row-major [m][k] storage, row stride in halves.
__device__ __forceinline__ v16h load_a_frag(const _Float16* base, int stride,
                                            int m0, int kb, int lane) {
  int m   = m0 + (lane & 15);
  int off = m * stride + kb + ((lane >> 4) << 4);
  v8h lo = *(const v8h*)(base + off);
  v8h hi = *(const v8h*)(base + off + 8);
  return combine16(lo, hi);
}

// B fragment (32x16 f16) from TRANSPOSED [n][k] storage, row stride in halves.
__device__ __forceinline__ v16h load_bt_frag(const _Float16* base, int stride,
                                             int n0, int kb, int lane) {
  int n   = n0 + (lane & 15);
  int off = n * stride + kb + ((lane >> 4) << 4);
  v8h lo = *(const v8h*)(base + off);
  v8h hi = *(const v8h*)(base + off + 8);
  return combine16(lo, hi);
}

// -------- K0: convert f32 weights to f16 once --------
__global__ void k_cvt(const float* __restrict__ wqkv, const float* __restrict__ wproj,
                      _Float16* __restrict__ wq16, _Float16* __restrict__ wp16) {
  int i = blockIdx.x * blockDim.x + threadIdx.x;
  if (i < 256 * 64) wq16[i] = (_Float16)wqkv[i];
  if (i < 64 * 64)  wp16[i] = (_Float16)wproj[i];
}

// -------- K1: per-pixel LayerNorm (c=64) + zero-map threshold --------
__global__ void k_ln(const float* __restrict__ x, const float* __restrict__ zmap,
                     const float* __restrict__ lnw, const float* __restrict__ lnb,
                     _Float16* __restrict__ xn, _Float16* __restrict__ zm) {
  int p = blockIdx.x * blockDim.x + threadIdx.x;
  if (p >= PTOT) return;
  int bi = p / HWPIX, hw = p % HWPIX;
  const float* xp = x + (size_t)bi * CCH * HWPIX + hw;
  float sum = 0.f, sumsq = 0.f;
  for (int c = 0; c < CCH; ++c) {
    float v = xp[(size_t)c * HWPIX];
    sum += v; sumsq += v * v;
  }
  float mu  = sum * (1.f / CCH);
  float var = sumsq * (1.f / CCH) - mu * mu;
  float inv = rsqrtf(var + 1e-5f);
  for (int c0 = 0; c0 < CCH; c0 += 8) {
    v8h t;
#pragma unroll
    for (int j = 0; j < 8; ++j) {
      int c = c0 + j;
      float v = xp[(size_t)c * HWPIX];
      t[j] = (_Float16)((v - mu) * inv * lnw[c] + lnb[c]);
    }
    *(v8h*)(xn + (size_t)p * CCH + c0) = t;
  }
  const float* zp = zmap + (size_t)bi * CCH * HWPIX + hw;
  for (int c = 0; c < CCH; ++c)
    zm[(size_t)c * PTOT + p] = (_Float16)(zp[(size_t)c * HWPIX] > 0.2f ? 1.f : 0.f);
}

// -------- K2: qkv 1x1 conv as WMMA GEMM: [256 x 64] * [64 x P] --------
__global__ __launch_bounds__(256)
void k_qkv(const _Float16* __restrict__ wq16, const _Float16* __restrict__ xn,
           _Float16* __restrict__ qkv) {
  int lane = threadIdx.x & 31, wv = threadIdx.x >> 5;
  int n0 = (blockIdx.x * 8 + wv) * 16;
  int m0 = blockIdx.y * 16;
  v8f acc = {};
#pragma unroll
  for (int kc = 0; kc < 2; ++kc) {
    int kb = kc * 32;
    v16h a = load_a_frag(wq16, 64, m0, kb, lane);
    v16h b = load_bt_frag(xn, 64, n0, kb, lane);
    acc = wmma32(a, b, acc);
  }
  int o = m0 + ((lane >> 4) << 3);
  int p = n0 + (lane & 15);
#pragma unroll
  for (int r = 0; r < 8; ++r)
    qkv[(size_t)(o + r) * PTOT + p] = (_Float16)acc[r];
}

// -------- K3: depthwise 3x3, groups=256, zero pad, 8-wide along x --------
__device__ __forceinline__ void load_row10(const _Float16* plane, int y, int x0,
                                           float* r) {
  const _Float16* rp = plane + y * WIMG + x0;
  v8h c = *(const v8h*)rp;
  r[0] = (x0 > 0) ? (float)rp[-1] : 0.f;
#pragma unroll
  for (int j = 0; j < 8; ++j) r[1 + j] = (float)c[j];
  r[9] = (x0 + 8 < WIMG) ? (float)rp[8] : 0.f;
}

__global__ void k_dw(const _Float16* __restrict__ qkv, const float* __restrict__ wdw,
                     _Float16* __restrict__ dst) {
  size_t idx = (size_t)blockIdx.x * blockDim.x + threadIdx.x;  // (o, p/8)
  int och = (int)(idx / (PTOT / 8));
  int p0  = (int)(idx % (PTOT / 8)) * 8;
  int bi = p0 / HWPIX, hw = p0 % HWPIX;
  int y = hw / WIMG, x0 = hw % WIMG;
  const float* wd = wdw + och * 9;
  const _Float16* plane = qkv + (size_t)och * PTOT + (size_t)bi * HWPIX;
  float r0[10], r1[10], r2[10];
#pragma unroll
  for (int j = 0; j < 10; ++j) { r0[j] = 0.f; r2[j] = 0.f; }
  if (y > 0)        load_row10(plane, y - 1, x0, r0);
  load_row10(plane, y, x0, r1);
  if (y < WIMG - 1) load_row10(plane, y + 1, x0, r2);
  v8h outv;
#pragma unroll
  for (int j = 0; j < 8; ++j) {
    float s = r0[j] * wd[0] + r0[j + 1] * wd[1] + r0[j + 2] * wd[2]
            + r1[j] * wd[3] + r1[j + 1] * wd[4] + r1[j + 2] * wd[5]
            + r2[j] * wd[6] + r2[j + 1] * wd[7] + r2[j + 2] * wd[8];
    outv[j] = (_Float16)s;
  }
  *(v8h*)(dst + (size_t)och * PTOT + p0) = outv;
}

// -------- K4: fused block attention for one (b, head, ch) group --------
// q,k,v planes are [c][PTOT] f16; out is pixel-major [p][64] f16.
__global__ __launch_bounds__(256)
void k_attn(const _Float16* __restrict__ qm, const _Float16* __restrict__ km,
            const _Float16* __restrict__ vm, const float* __restrict__ temp,
            _Float16* __restrict__ outp, int do_norm) {
  __shared__ __attribute__((aligned(32))) _Float16 qs[TT * FC];   // 16 KB
  __shared__ __attribute__((aligned(32))) _Float16 ks[TT * FC];   // 16 KB
  __shared__ __attribute__((aligned(32))) float    Sbuf[TT * TT]; // 16 KB
  __shared__ __attribute__((aligned(32))) _Float16 abuf[TT * TT]; // 8 KB
  __shared__ float nq[TT], nk[TT];

  int g  = blockIdx.x;       // 0..127
  int bi = g >> 6;
  int cg = g & 63;           // == head*16 + ch
  int head = cg >> 4;
  size_t plane = (size_t)cg * PTOT + (size_t)bi * HWPIX;
  int tid = threadIdx.x, lane = tid & 31, wv = tid >> 5;
  int lhi = lane >> 4, llo = lane & 15;

  // spatial offset of block-row l = (f_local, n1) in chunk fcb
  auto sp_of = [&](int fcb, int l) {
    int fl = l >> 3, n1 = l & 7;
    int f = fcb * FC + fl;
    return ((f / 48) * 8 + n1) * WIMG + (f % 48) * 8;
  };

  // ---- Phase A: S = q * k^T (64x64, K=2304), register-prefetch pipeline ----
  v8h pq[4], pk[4];
  auto prefetchA = [&](int fcb) {
#pragma unroll
    for (int it = 0; it < 4; ++it) {
      int sp = sp_of(fcb, tid + it * 256);
      pq[it] = *(const v8h*)(qm + plane + sp);
      pk[it] = *(const v8h*)(km + plane + sp);
    }
  };
  prefetchA(0);

  v8f acc0 = {}, acc1 = {};
  for (int fcb = 0; fcb < NCHUNK; ++fcb) {
    // spill prefetched registers into LDS tiles [t][f]
#pragma unroll
    for (int it = 0; it < 4; ++it) {
      int l = tid + it * 256;
      int fl = l >> 3, n1 = l & 7;
#pragma unroll
      for (int n2 = 0; n2 < 8; ++n2) {
        int t = n1 * 8 + n2;
        qs[t * FC + fl] = pq[it][n2];
        ks[t * FC + fl] = pk[it][n2];
      }
    }
    __syncthreads();
    if (fcb + 1 < NCHUNK) prefetchA(fcb + 1);  // overlap with compute below
    if (do_norm && tid < 128) {
      int t = tid & 63;
      const _Float16* src = (tid < 64) ? (qs + t * FC) : (ks + t * FC);
      float s = 0.f;
      for (int i = 0; i < FC; ++i) { float v = (float)src[i]; s += v * v; }
      float* dst = (tid < 64) ? nq : nk;
      if (fcb == 0) dst[t] = s; else dst[t] += s;
    }
    {
      int tile0 = wv * 2;
      int ti0 = tile0 >> 2, si0 = tile0 & 3;
      int ti1 = (tile0 + 1) >> 2, si1 = (tile0 + 1) & 3;
#pragma unroll
      for (int kc = 0; kc < FC / 32; ++kc) {
        int kb = kc * 32;
        v16h a0 = load_a_frag(qs, FC, ti0 * 16, kb, lane);
        v16h b0 = load_bt_frag(ks, FC, si0 * 16, kb, lane);
        acc0 = wmma32(a0, b0, acc0);
        v16h a1 = load_a_frag(qs, FC, ti1 * 16, kb, lane);
        v16h b1 = load_bt_frag(ks, FC, si1 * 16, kb, lane);
        acc1 = wmma32(a1, b1, acc1);
      }
    }
    __syncthreads();
  }
  // spill S tiles to LDS
  {
    int tile0 = wv * 2;
#pragma unroll
    for (int rep = 0; rep < 2; ++rep) {
      int tile = tile0 + rep;
      int ti = tile >> 2, si = tile & 3;
      v8f a = rep ? acc1 : acc0;
#pragma unroll
      for (int r = 0; r < 8; ++r)
        Sbuf[(ti * 16 + (lhi << 3) + r) * TT + si * 16 + llo] = a[r];
    }
  }
  __syncthreads();

  // ---- Phase B: scale (l2 norms + temperature) + softmax, rows of 64 ----
  float tv = temp[head];
  if (tid < TT) {
    int t = tid;
    float qn = do_norm ? fmaxf(sqrtf(nq[t]), 1e-12f) : 1.f;
    float mx = -1e30f;
    for (int s = 0; s < TT; ++s) {
      float v = Sbuf[t * TT + s];
      if (do_norm) v /= (qn * fmaxf(sqrtf(nk[s]), 1e-12f));
      v *= tv;
      Sbuf[t * TT + s] = v;
      mx = fmaxf(mx, v);
    }
    float sum = 0.f;
    for (int s = 0; s < TT; ++s) {
      float e = __expf(Sbuf[t * TT + s] - mx);
      Sbuf[t * TT + s] = e;
      sum += e;
    }
    float inv = 1.f / sum;
    for (int s = 0; s < TT; ++s)
      abuf[t * TT + s] = (_Float16)(Sbuf[t * TT + s] * inv);
  }
  __syncthreads();

  // ---- Phase C: out = attn(64x64) * v(64x2304) ----
  // v chunks staged TRANSPOSED (vs[f][s], contiguous b128 in LDS) through the
  // async memory->LDS path, double-buffered on ASYNCcnt (4 loads/wave/chunk).
  // Ping-pong buffer selected by ternary (no LDS-pointer array: lld can't
  // relocate addrspacecast in static initializers).
  auto stage_async = [&](int fcb, _Float16* buf) {
#pragma unroll
    for (int it = 0; it < 4; ++it) {
      int l = tid + it * 256;
      int fl = l >> 3, n1 = l & 7;
      int sp = sp_of(fcb, l);
      unsigned lds = (unsigned)(size_t)&buf[fl * TT + n1 * 8];
      unsigned long long ga = (unsigned long long)(size_t)(vm + plane + sp);
      asm volatile("global_load_async_to_lds_b128 %0, %1, off"
                   :: "v"(lds), "v"(ga) : "memory");
    }
  };

  stage_async(0, qs);
  for (int fcb = 0; fcb < NCHUNK; ++fcb) {
    if (fcb + 1 < NCHUNK) {
      stage_async(fcb + 1, ((fcb + 1) & 1) ? ks : qs);
      asm volatile("s_wait_asynccnt 0x4" ::: "memory");  // chunk fcb landed
    } else {
      asm volatile("s_wait_asynccnt 0x0" ::: "memory");
    }
    __syncthreads();
    const _Float16* vs = (fcb & 1) ? ks : qs;
#pragma unroll
    for (int rep = 0; rep < 4; ++rep) {
      int tile = wv * 4 + rep;   // 32 tiles: 4 t-tiles x 8 f-tiles
      int ti = tile >> 3;
      int fi = tile & 7;
      v8f acc = {};
#pragma unroll
      for (int kc = 0; kc < 2; ++kc) {
        int kb = kc * 32;
        v16h a = load_a_frag(abuf, TT, ti * 16, kb, lane);
        v16h b = load_bt_frag(vs, TT, fi * 16, kb, lane);
        acc = wmma32(a, b, acc);
      }
      int fl = fi * 16 + llo;
      int f  = fcb * FC + fl;
      int g1 = f / 48, g2 = f % 48;
#pragma unroll
      for (int r = 0; r < 8; ++r) {
        int t = ti * 16 + (lhi << 3) + r;
        int y = g1 * 8 + (t >> 3);
        int x = g2 * 8 + (t & 7);
        outp[((size_t)bi * HWPIX + y * WIMG + x) * CCH + cg] = (_Float16)acc[r];
      }
    }
    __syncthreads();   // protects ping-pong buffer reuse two iterations out
  }
}

// -------- K5: result = w_proj * (outA + outZ) + 2*x --------
__global__ __launch_bounds__(256)
void k_proj(const _Float16* __restrict__ wp16, const _Float16* __restrict__ oa,
            const _Float16* __restrict__ oz, const float* __restrict__ xin,
            float* __restrict__ out) {
  int lane = threadIdx.x & 31, wv = threadIdx.x >> 5;
  int n0 = (blockIdx.x * 8 + wv) * 16;
  int m0 = blockIdx.y * 16;
  v8f acc = {};
#pragma unroll
  for (int kc = 0; kc < 2; ++kc) {
    int kb = kc * 32;
    v16h a = load_a_frag(wp16, 64, m0, kb, lane);
    int n = n0 + (lane & 15);
    int off = n * 64 + kb + ((lane >> 4) << 4);
    v8h lo = *(const v8h*)(oa + off)     + *(const v8h*)(oz + off);
    v8h hi = *(const v8h*)(oa + off + 8) + *(const v8h*)(oz + off + 8);
    acc = wmma32(a, combine16(lo, hi), acc);
  }
  int p = n0 + (lane & 15);
  int bi = p / HWPIX, hw = p % HWPIX;
  int o = m0 + ((lane >> 4) << 3);
#pragma unroll
  for (int r = 0; r < 8; ++r) {
    size_t idx = ((size_t)bi * CCH + o + r) * HWPIX + hw;
    out[idx] = acc[r] + 2.f * xin[idx];
  }
}

extern "C" void kernel_launch(void* const* d_in, const int* in_sizes, int n_in,
                              void* d_out, int out_size, void* d_ws, size_t ws_size,
                              hipStream_t stream) {
  const float* x     = (const float*)d_in[0];
  const float* zmap  = (const float*)d_in[1];
  const float* lnw   = (const float*)d_in[2];
  const float* lnb   = (const float*)d_in[3];
  const float* wqkv  = (const float*)d_in[4];
  const float* wdw   = (const float*)d_in[5];
  const float* wproj = (const float*)d_in[6];
  const float* temp1 = (const float*)d_in[7];
  const float* temp2 = (const float*)d_in[8];
  float* out = (float*)d_out;

  const size_t SXN = (size_t)PTOT * CCH * 2;  // one f16 [64][P] tensor: 37.75 MB
  char* ws = (char*)d_ws;
  _Float16* xn   = (_Float16*)(ws);                       // [p][64]
  _Float16* zm   = (_Float16*)(ws + SXN);                 // [c][P]
  _Float16* qkv  = (_Float16*)(ws + 2 * SXN);             // [256][P]
  _Float16* dw   = (_Float16*)(ws + 6 * SXN);             // [256][P]
  _Float16* outA = (_Float16*)(ws + 10 * SXN);            // [p][64]
  _Float16* outZ = (_Float16*)(ws + 11 * SXN);            // [p][64]
  _Float16* wq16 = (_Float16*)(ws + 12 * SXN);            // 32 KB
  _Float16* wp16 = wq16 + 256 * 64;                       // 8 KB

  k_cvt<<<(256 * 64 + 255) / 256, 256, 0, stream>>>(wqkv, wproj, wq16, wp16);
  k_ln<<<PTOT / 256, 256, 0, stream>>>(x, zmap, lnw, lnb, xn, zm);
  k_qkv<<<dim3(PTOT / 128, 16), 256, 0, stream>>>(wq16, xn, qkv);
  k_dw<<<(unsigned)(256ull * PTOT / 8 / 256), 256, 0, stream>>>(qkv, wdw, dw);

  // main path: q,k,v = dw channel sections 0,64,128; l2norm on
  k_attn<<<128, 256, 0, stream>>>(dw, dw + (size_t)64 * PTOT,
                                  dw + (size_t)128 * PTOT, temp1, outA, 1);
  // zero-map path: q=k=zm, v = v1 (section 192); no l2norm
  k_attn<<<128, 256, 0, stream>>>(zm, zm, dw + (size_t)192 * PTOT, temp2, outZ, 0);

  k_proj<<<dim3(PTOT / 128, 4), 256, 0, stream>>>(wp16, outA, outZ, x, out);
}
